// GraphTransformer_68917045231692
// MI455X (gfx1250) — matrix operand order repaired and checked by reading, not compiled, not run on hardware
//
#include <hip/hip_runtime.h>
#include <hip/hip_bf16.h>

// GraphTransformer (2x PyG TransformerConv) for MI455X / gfx1250, wave32.
// N=50000 nodes, E=1600000 edges, IN=128, HID=32, H=4 -> HC=128, OUT=64, ED=64.
//
// Memory-bound (23.3 TB/s HBM, 192MB L2): node tensors (25MB) stay L2-resident
// so edge gathers are L2 hits; dominant HBM streams are edge_feats (410MB/layer)
// and the e-projection intermediate (819MB). GEMMs on v_wmma_f32_16x16x32_f16:
//   - weights pre-packed once (~220KB) into per-lane B-fragment f16 layout
//     -> one 32B contiguous load per lane per K-step,
//   - 16-row A tile staged per block into LDS via GLOBAL_LOAD_ASYNC_TO_LDS_B128
//     (ASYNCcnt path; falls back to plain copy if builtin missing),
//   - K templated so the WMMA loop fully unrolls (4 wmma back-to-back).
// Segment softmax: 3 edge passes, global_atomic_max_u32 (order-preserving float
// encoding) + global_atomic_add_f32; skip connections folded into accumulators.

typedef __attribute__((ext_vector_type(16))) _Float16 v16h;
typedef __attribute__((ext_vector_type(8)))  float    v8f;
typedef int v4i __attribute__((vector_size(16)));   // matches async-LDS builtin proto

#define AS1 __attribute__((address_space(1)))
#define AS3 __attribute__((address_space(3)))

#if defined(__has_builtin)
#if __has_builtin(__builtin_amdgcn_global_load_async_to_lds_b128)
#define HAVE_ASYNC_LDS 1
#endif
#endif
#ifndef HAVE_ASYNC_LDS
#define HAVE_ASYNC_LDS 0
#endif

// ---- order-preserving float <-> uint encoding for atomic max ----------------
__device__ __forceinline__ unsigned fenc(float f) {
  unsigned u = __float_as_uint(f);
  return ((int)u >= 0) ? (u | 0x80000000u) : ~u;
}
__device__ __forceinline__ float fdec(unsigned e) {
  return __uint_as_float((e & 0x80000000u) ? (e & 0x7fffffffu) : ~e);
}

// ---- one-time weight pack: B[K x ncols] f32 -> per-lane WMMA B fragments ----
// Fragment layout chosen here (contiguous k, same order as the A-fragment; wave32): lane half hf, col c:
//   b[j] = B[kb + 16*hf + j][c],  j = 0..15
// Packed address: Bp[(((kt*ncols + c)*2 + hf)*16) + j]  (kt = kb/32)
__global__ void pack_weights_f16(const float* __restrict__ B,
                                 _Float16* __restrict__ Bp, int ncols, long total) {
  long i = (long)blockIdx.x * blockDim.x + threadIdx.x;
  if (i >= total) return;
  const int  j  = (int)(i & 15);
  long r        = i >> 4;
  const int  hf = (int)(r & 1);  r >>= 1;
  const int  c  = (int)(r % ncols);
  const int  t  = (int)(r / ncols);
  Bp[i] = (_Float16)B[(long)(t * 32 + hf * 16 + j) * ncols + c];
}

// ---- WMMA GEMM: C[16 x NCOLS] per block = A_tile[16 x K] * Bp (+bias) -------
// blockDim = 2*NCOLS (NCOLS/16 waves); blockIdx.x = row tile; wave = col tile.
// A-fragment: lane half hf, row rc:
//   a[0..15] = K(kb + hf*16 + i)
// C/D: c[r] = C[row = r + 8*hf][col = lane&15]
template <int K, int NCOLS>
__global__ void gemm_wmma(const float* __restrict__ A,
                          const _Float16* __restrict__ Bp,
                          const float* __restrict__ bias,
                          float* __restrict__ C) {
  __shared__ float As[16 * K];             // raw f32 tile (max 8KB)

  const int  tid  = threadIdx.x;
  const int  lane = tid & 31;
  const int  wave = tid >> 5;
  const int  hf   = lane >> 4;
  const int  rc   = lane & 15;
  const long rowTile = (long)blockIdx.x * 16;
  constexpr int NTHREADS = 2 * NCOLS;

  const float* Ag = A + rowTile * K;
  __builtin_prefetch(Ag + 16 * K, 0, 1);   // next row tile -> global_prefetch

#if HAVE_ASYNC_LDS
  // async global->LDS staging, 16B per transfer, no VGPR round trip (ASYNCcnt)
  constexpr int TPT = (16 * K) / (4 * NTHREADS);   // b128 transfers per thread
#pragma unroll
  for (int i = 0; i < TPT; ++i) {
    const int off = (tid + i * NTHREADS) * 4;      // floats
    __builtin_amdgcn_global_load_async_to_lds_b128(
        (AS1 v4i*)(Ag + off), (AS3 v4i*)(As + off), 0, 0);
  }
  asm volatile("s_wait_asynccnt 0" ::: "memory");
#else
  constexpr int FPT = (16 * K) / NTHREADS;         // floats per thread
  {
    const int f = tid * FPT;
#pragma unroll
    for (int i = 0; i < FPT; ++i) As[f + i] = Ag[f + i];
  }
#endif
  __syncthreads();

  const int n0 = wave * 16;
  const float* arow = As + rc * K;
  v8f acc = {};
#pragma unroll
  for (int kt = 0; kt < K / 32; ++kt) {
    const int kb = kt * 32;
    const v8f lo = *(const v8f*)(arow + kb + hf * 16);       // 2x ds_load_b128
    const v8f hi = *(const v8f*)(arow + kb + hf * 16 + 8);   // 2x ds_load_b128
    v16h af;
#pragma unroll
    for (int i = 0; i < 8; ++i) {
      af[i]     = (_Float16)lo[i];
      af[i + 8] = (_Float16)hi[i];
    }
    const v16h bf = *(const v16h*)(Bp + ((long)(kt * NCOLS + n0 + rc) * 2 + hf) * 16);
    acc = __builtin_amdgcn_wmma_f32_16x16x32_f16(false, af, false, bf,
                                                 (short)0, acc, false, false);
  }
  const float bv = bias ? bias[n0 + rc] : 0.0f;
#pragma unroll
  for (int r = 0; r < 8; ++r)
    C[(rowTile + r + hf * 8) * (long)NCOLS + n0 + rc] = acc[r] + bv;
}

// ---- init segment-max (encoded -inf == 0) and segment-sum buffers -----------
__global__ void init_md(unsigned* __restrict__ m, float* __restrict__ den, long n) {
  long i = (long)blockIdx.x * blockDim.x + threadIdx.x;
  if (i < n) { m[i] = 0u; den[i] = 0.0f; }
}

// ---- pass 1: alpha[e,h] = dot(q[dst],k[src]+e_proj)/sqrt(ch); seg-max -------
template <int HEADS, int CPL>
__global__ void edge_scores(const float* __restrict__ q,
                            const float* __restrict__ k,
                            const float* __restrict__ ep,
                            const int* __restrict__ src,
                            const int* __restrict__ dst,
                            float* __restrict__ alpha,
                            unsigned* __restrict__ mbuf, long E) {
  const int  lane = threadIdx.x & 31;
  const long edge = (long)blockIdx.x * (blockDim.x >> 5) + (threadIdx.x >> 5);
  if (edge >= E) return;
  constexpr int CH  = 32 * CPL;
  constexpr int LPH = 32 / HEADS;           // lanes per head
  const int s = src[edge], d = dst[edge];
  const float* qp = q  + (long)d * CH + lane * CPL;
  const float* kp = k  + (long)s * CH + lane * CPL;
  const float* xp = ep + edge * (long)CH + lane * CPL;
  float sum = 0.0f;
#pragma unroll
  for (int i = 0; i < CPL; ++i) sum += qp[i] * (kp[i] + xp[i]);
#pragma unroll
  for (int off = 1; off < LPH; off <<= 1) sum += __shfl_xor(sum, off, 32);
  if ((lane & (LPH - 1)) == 0) {
    const int h = lane / LPH;
    const float a = sum * rsqrtf((float)(CH / HEADS));
    alpha[edge * HEADS + h] = a;
    atomicMax(&mbuf[(long)d * HEADS + h], fenc(a));
  }
}

// ---- pass 2: a = exp(alpha - m[dst]); den[dst] += a -------------------------
template <int HEADS>
__global__ void edge_expsum(const int* __restrict__ dst,
                            float* __restrict__ alpha,
                            const unsigned* __restrict__ mbuf,
                            float* __restrict__ den, long E) {
  long i = (long)blockIdx.x * blockDim.x + threadIdx.x;   // over E*HEADS
  if (i >= E * HEADS) return;
  const long e = i / HEADS;
  const int  h = (int)(i - e * HEADS);
  const int  d = dst[e];
  const float m = fdec(mbuf[(long)d * HEADS + h]);
  const float a = __expf(alpha[i] - m);
  alpha[i] = a;
  atomicAdd(&den[(long)d * HEADS + h], a);
}

// ---- pass 3: out[dst] += (alpha/den[dst]) * (v[src] + e_proj) ---------------
template <int HEADS, int CPL>
__global__ void edge_scatter(const float* __restrict__ v,
                             const float* __restrict__ ep,
                             const float* __restrict__ alpha,
                             const float* __restrict__ den,
                             const int* __restrict__ src,
                             const int* __restrict__ dst,
                             float* __restrict__ out, long E) {
  const int  lane = threadIdx.x & 31;
  const long edge = (long)blockIdx.x * (blockDim.x >> 5) + (threadIdx.x >> 5);
  if (edge >= E) return;
  constexpr int CH  = 32 * CPL;
  constexpr int LPH = 32 / HEADS;
  const int s = src[edge], d = dst[edge];
  const int h = lane / LPH;
  const float coef = alpha[edge * HEADS + h] / den[(long)d * HEADS + h];
  const float* vp = v  + (long)s * CH + lane * CPL;
  const float* xp = ep + edge * (long)CH + lane * CPL;
  float* op = out + (long)d * CH + lane * CPL;
#pragma unroll
  for (int i = 0; i < CPL; ++i)
    atomicAdd(&op[i], coef * (vp[i] + xp[i]));   // global_atomic_add_f32
}

// ---- relu ------------------------------------------------------------------
__global__ void relu_inplace(float* __restrict__ x, long n) {
  long i = (long)blockIdx.x * blockDim.x + threadIdx.x;
  if (i < n) x[i] = fmaxf(x[i], 0.0f);
}

extern "C" void kernel_launch(void* const* d_in, const int* in_sizes, int n_in,
                              void* d_out, int out_size, void* d_ws, size_t ws_size,
                              hipStream_t stream) {
  (void)n_in; (void)out_size; (void)ws_size;
  const float* x   = (const float*)d_in[0];
  const float* ef  = (const float*)d_in[1];
  const int*   ei  = (const int*)  d_in[2];
  const float* Wq1 = (const float*)d_in[3];  const float* bq1 = (const float*)d_in[4];
  const float* Wk1 = (const float*)d_in[5];  const float* bk1 = (const float*)d_in[6];
  const float* Wv1 = (const float*)d_in[7];  const float* bv1 = (const float*)d_in[8];
  const float* We1 = (const float*)d_in[9];
  const float* Ws1 = (const float*)d_in[10]; const float* bs1 = (const float*)d_in[11];
  const float* Wq2 = (const float*)d_in[12]; const float* bq2 = (const float*)d_in[13];
  const float* Wk2 = (const float*)d_in[14]; const float* bk2 = (const float*)d_in[15];
  const float* Wv2 = (const float*)d_in[16]; const float* bv2 = (const float*)d_in[17];
  const float* We2 = (const float*)d_in[18];
  const float* Ws2 = (const float*)d_in[19]; const float* bs2 = (const float*)d_in[20];

  const long N = in_sizes[0] / 128;      // 50000
  const long E = in_sizes[1] / 64;       // 1600000
  const int* src = ei;
  const int* dst = ei + E;
  float* out = (float*)d_out;            // N x 64

  // ---- workspace carve: packed f16 weights first (all sizes 64B-aligned) ----
  _Float16* wp = (_Float16*)d_ws;
  _Float16* pWq1 = wp; wp += 16384;      // 128x128
  _Float16* pWk1 = wp; wp += 16384;
  _Float16* pWv1 = wp; wp += 16384;
  _Float16* pWs1 = wp; wp += 16384;
  _Float16* pWe1 = wp; wp += 8192;       // 64x128
  _Float16* pWq2 = wp; wp += 8192;       // 128x64
  _Float16* pWk2 = wp; wp += 8192;
  _Float16* pWv2 = wp; wp += 8192;
  _Float16* pWs2 = wp; wp += 8192;
  _Float16* pWe2 = wp; wp += 4096;       // 64x64

  float* q1 = (float*)wp;                // N*128
  float* k1 = q1 + N * 128;              // N*128
  float* v1 = k1 + N * 128;              // N*128
  float* h  = v1 + N * 128;              // N*128 (layer-1 output accumulator)
  float* q2 = h  + N * 128;              // N*64
  float* k2 = q2 + N * 64;               // N*64
  float* v2 = k2 + N * 64;               // N*64
  float* e1 = v2 + N * 64;               // E*128 (layer 1), reused E*64 (layer 2)
  float* alpha = e1 + E * 128;           // E*4, reused E*1
  unsigned* mb = (unsigned*)(alpha + E * 4);   // N*4, reused N
  float* den = (float*)(mb + N * 4);           // N*4, reused N

  const dim3 blk256(256), blk128(128);
  const int nTiles = (int)(N / 16);      // exact: 3125
  const int eTiles = (int)(E / 16);      // exact: 100000
  const int eWaveB = (int)((E + 7) / 8); // 8 edge-waves per 256-thread block

  // ---- pack all weight matrices into WMMA B-fragment layout (f16) ----------
  pack_weights_f16<<<64, blk256, 0, stream>>>(Wq1, pWq1, 128, 16384);
  pack_weights_f16<<<64, blk256, 0, stream>>>(Wk1, pWk1, 128, 16384);
  pack_weights_f16<<<64, blk256, 0, stream>>>(Wv1, pWv1, 128, 16384);
  pack_weights_f16<<<64, blk256, 0, stream>>>(Ws1, pWs1, 128, 16384);
  pack_weights_f16<<<32, blk256, 0, stream>>>(We1, pWe1, 128, 8192);
  pack_weights_f16<<<32, blk256, 0, stream>>>(Wq2, pWq2, 64, 8192);
  pack_weights_f16<<<32, blk256, 0, stream>>>(Wk2, pWk2, 64, 8192);
  pack_weights_f16<<<32, blk256, 0, stream>>>(Wv2, pWv2, 64, 8192);
  pack_weights_f16<<<32, blk256, 0, stream>>>(Ws2, pWs2, 64, 8192);
  pack_weights_f16<<<16, blk256, 0, stream>>>(We2, pWe2, 64, 4096);

  // ---------------- layer 1: IN=128 -> H=4 x HID=32 (HC=128) ----------------
  gemm_wmma<128, 128><<<nTiles, blk256, 0, stream>>>(x, pWq1, bq1, q1);
  gemm_wmma<128, 128><<<nTiles, blk256, 0, stream>>>(x, pWk1, bk1, k1);
  gemm_wmma<128, 128><<<nTiles, blk256, 0, stream>>>(x, pWv1, bv1, v1);
  gemm_wmma<128, 128><<<nTiles, blk256, 0, stream>>>(x, pWs1, bs1, h);   // skip
  gemm_wmma< 64, 128><<<eTiles, blk256, 0, stream>>>(ef, pWe1, nullptr, e1);

  init_md<<<(unsigned)((N * 4 + 255) / 256), blk256, 0, stream>>>(mb, den, N * 4);
  edge_scores<4, 4><<<eWaveB, blk256, 0, stream>>>(q1, k1, e1, src, dst, alpha, mb, E);
  edge_expsum<4><<<(unsigned)((E * 4 + 255) / 256), blk256, 0, stream>>>(dst, alpha, mb, den, E);
  edge_scatter<4, 4><<<eWaveB, blk256, 0, stream>>>(v1, e1, alpha, den, src, dst, h, E);
  relu_inplace<<<(unsigned)((N * 128 + 255) / 256), blk256, 0, stream>>>(h, N * 128);

  // ---------------- layer 2: HC=128 -> OUT=64, heads=1 ----------------------
  gemm_wmma<128, 64><<<nTiles, blk128, 0, stream>>>(h, pWq2, bq2, q2);
  gemm_wmma<128, 64><<<nTiles, blk128, 0, stream>>>(h, pWk2, bk2, k2);
  gemm_wmma<128, 64><<<nTiles, blk128, 0, stream>>>(h, pWv2, bv2, v2);
  gemm_wmma<128, 64><<<nTiles, blk128, 0, stream>>>(h, pWs2, bs2, out);  // skip
  gemm_wmma< 64, 64><<<eTiles, blk128, 0, stream>>>(ef, pWe2, nullptr, e1);

  init_md<<<(unsigned)((N + 255) / 256), blk256, 0, stream>>>(mb, den, N);
  edge_scores<1, 2><<<eWaveB, blk256, 0, stream>>>(q2, k2, e1, src, dst, alpha, mb, E);
  edge_expsum<1><<<(unsigned)((E + 255) / 256), blk256, 0, stream>>>(dst, alpha, mb, den, E);
  edge_scatter<1, 2><<<eWaveB, blk256, 0, stream>>>(v2, e1, alpha, den, src, dst, out, E);
}
